// Block_30605936951303
// MI455X (gfx1250) — matrix operand order repaired and checked by reading, not compiled, not run on hardware
//
#include <hip/hip_runtime.h>
#include <hip/hip_bf16.h>
#include <math.h>

typedef _Float16 half_t;
typedef __attribute__((ext_vector_type(8)))  _Float16 half8;
typedef __attribute__((ext_vector_type(16))) _Float16 v16h;
typedef __attribute__((ext_vector_type(8)))  float    v8f;

#define BATCH     8
#define HIMG      56
#define WIMG      56
#define SEQ       (HIMG*WIMG)           // 3136
#define DIMC      768
#define HEADS     12
#define HEAD_DIM  64
#define WS        14
#define NWIN_SIDE (HIMG/WS)             // 4
#define NWIN      (NWIN_SIDE*NWIN_SIDE) // 16
#define BW        (BATCH*NWIN)          // 128 windows
#define NW        (WS*WS)               // 196 tokens / window
#define NKPAD     224                   // keys padded to 7*32 for WMMA K-steps
#define MROWS     (BATCH*SEQ)           // 25088
#define MLP_HID   (4*DIMC)              // 3072
#define ATT_SCALE 0.125f                // 64^-0.5

// ---------------------------------------------------------------------------
// CDNA5 async-to-LDS DMA (ASYNCcnt-tracked) + WMMA helpers
// ---------------------------------------------------------------------------
__device__ __forceinline__ void async_load_b128(unsigned lds_off, const void* gptr) {
  // GLOBAL_LOAD_ASYNC_TO_LDS_B128: vdst = LDS byte offset, vaddr = 64b address
  asm volatile("global_load_async_to_lds_b128 %0, %1, off"
               :: "v"(lds_off), "v"(gptr) : "memory");
}
__device__ __forceinline__ void wait_async_le1() {
  asm volatile("s_wait_asynccnt 0x1" ::: "memory");
}
__device__ __forceinline__ void wait_async_le0() {
  asm volatile("s_wait_asynccnt 0x0" ::: "memory");
}

__device__ __forceinline__ v8f wmma16(v16h a, v16h b, v8f c) {
  return __builtin_amdgcn_wmma_f32_16x16x32_f16(false, a, false, b, (short)0, c,
                                                false, false);
}

// A (16x32 f16, MxK): lane L -> row M = L&15; 16 contiguous K at (L>>4)*16.
__device__ __forceinline__ v16h load_frag_a(const half_t* base, int ldk,
                                            int lane, int k0) {
  const half_t* p = base + (size_t)(lane & 15) * ldk + k0 + ((lane >> 4) << 4);
  half8 lo = *(const half8*)p;
  half8 hi = *(const half8*)(p + 8);
  return __builtin_shufflevector(lo, hi, 0,1,2,3,4,5,6,7,8,9,10,11,12,13,14,15);
}

// B (32x16 f16, KxN) from packed fragment storage: lane's 16 values contiguous.
__device__ __forceinline__ v16h load_frag_b_packed(const half_t* __restrict__ tile,
                                                   int lane) {
  const half_t* p = tile + lane * 16;
  half8 lo = *(const half8*)p;
  half8 hi = *(const half8*)(p + 8);
  return __builtin_shufflevector(lo, hi, 0,1,2,3,4,5,6,7,8,9,10,11,12,13,14,15);
}

// B (32x16, KxN), element (k,n) at base + n*ld + k (K contiguous per column).
__device__ __forceinline__ v16h load_frag_b_cols(const half_t* __restrict__ base,
                                                 int ld, int lane, int n0, int k0) {
  const half_t* p = base + (size_t)(n0 + (lane & 15)) * ld + k0 + ((lane >> 4) << 4);
  half8 lo = *(const half8*)p;
  half8 hi = *(const half8*)(p + 8);
  return __builtin_shufflevector(lo, hi, 0,1,2,3,4,5,6,7,8,9,10,11,12,13,14,15);
}

// ---------------------------------------------------------------------------
// Weight packing: f32 [K,N] row-major -> f16 WMMA B-fragment layout.
// ---------------------------------------------------------------------------
__global__ __launch_bounds__(256) void pack_b(const float* __restrict__ W,
                                              half_t* __restrict__ P,
                                              int K, int N) {
  size_t idx = (size_t)blockIdx.x * 256 + threadIdx.x;
  size_t total = (size_t)K * N;
  if (idx >= total) return;
  int t = (int)(idx >> 9);
  int r = (int)(idx & 511);
  int L = r >> 4;
  int q = r & 15;
  int ntiles = N >> 4;
  int ktile = t / ntiles, ntile = t - ktile * ntiles;
  int k = ktile * 32 + ((L >> 4) << 4) + q;
  int n = ntile * 16 + (L & 15);
  P[idx] = (half_t)W[(size_t)k * N + n];
}

// ---------------------------------------------------------------------------
// LayerNorm: f32 [rows, 768] -> f16, one workgroup per row.
// ---------------------------------------------------------------------------
__global__ __launch_bounds__(256) void ln_kernel(const float* __restrict__ in,
                                                 const float* __restrict__ sc,
                                                 const float* __restrict__ bi,
                                                 half_t* __restrict__ out) {
  __shared__ float red[256];
  const int row = blockIdx.x;
  const float* p = in + (size_t)row * DIMC;
  float v0 = p[threadIdx.x];
  float v1 = p[threadIdx.x + 256];
  float v2 = p[threadIdx.x + 512];
  red[threadIdx.x] = v0 + v1 + v2;
  __syncthreads();
  for (int off = 128; off > 0; off >>= 1) {
    if (threadIdx.x < off) red[threadIdx.x] += red[threadIdx.x + off];
    __syncthreads();
  }
  float mu = red[0] * (1.0f / DIMC);
  __syncthreads();
  float d0 = v0 - mu, d1 = v1 - mu, d2 = v2 - mu;
  red[threadIdx.x] = d0 * d0 + d1 * d1 + d2 * d2;
  __syncthreads();
  for (int off = 128; off > 0; off >>= 1) {
    if (threadIdx.x < off) red[threadIdx.x] += red[threadIdx.x + off];
    __syncthreads();
  }
  float inv = rsqrtf(red[0] * (1.0f / DIMC) + 1e-5f);
  half_t* o = out + (size_t)row * DIMC;
  o[threadIdx.x]       = (half_t)(d0 * inv * sc[threadIdx.x]       + bi[threadIdx.x]);
  o[threadIdx.x + 256] = (half_t)(d1 * inv * sc[threadIdx.x + 256] + bi[threadIdx.x + 256]);
  o[threadIdx.x + 512] = (half_t)(d2 * inv * sc[threadIdx.x + 512] + bi[threadIdx.x + 512]);
}

// ---------------------------------------------------------------------------
// WMMA GEMM with async-LDS-staged A (double buffered), C = A[M,K]*Bp[K,N].
// 256 threads = 8 waves (2x4), workgroup tile 64x256, wave tile 32x64
// (2x4 WMMA accumulators = 8 WMMAs per k-step per wave).
// MODE 0: qkv scatter; 1: proj+resid f32; 2: fc1+GELU f16; 3: fc2+resid f32.
// ---------------------------------------------------------------------------
template <int MODE>
__global__ __launch_bounds__(256) void gemm_ep(const half_t* __restrict__ A,
                                               const half_t* __restrict__ Bp,
                                               const float* __restrict__ bias,
                                               const float* __restrict__ resid,
                                               void* out0, void* out1, void* out2,
                                               int M, int N, int K) {
  __shared__ half_t tileA[2][64 * 32];   // double-buffered 64x32 A tile (8 KB)
  const int tid  = threadIdx.x;
  const int lane = tid & 31;
  const int wave = tid >> 5;
  const int wm = wave >> 2, wn = wave & 3;
  const int mblk = blockIdx.y * 64;
  const int m0 = mblk + wm * 32;
  const int n0 = blockIdx.x * 256 + wn * 64;
  const int ntiles = N >> 4;

  // Each thread DMAs one 16B chunk per stage: row = tid/4, col = (tid%4)*8.
  const int arow = tid >> 2;
  const int acol = (tid & 3) << 3;
  const half_t* agp = A + (size_t)(mblk + arow) * K + acol;
  const unsigned lds0 = (unsigned)(uintptr_t)&tileA[0][arow * 32 + acol];
  const unsigned lds1 = (unsigned)(uintptr_t)&tileA[1][arow * 32 + acol];

  async_load_b128(lds0, agp);            // stage 0
  async_load_b128(lds1, agp + 32);       // stage 1 (K >= 64 always here)

  v8f acc[2][4] = {};
#pragma clang loop unroll(disable)
  for (int k0 = 0; k0 < K; k0 += 32) {
    const int buf = (k0 >> 5) & 1;
    if (k0 + 32 < K) wait_async_le1();   // current stage complete
    else             wait_async_le0();
    __syncthreads();                     // LDS visible to all waves

    const half_t* btile = Bp + ((size_t)(k0 >> 5) * ntiles + (n0 >> 4)) * 512;
    __builtin_prefetch(btile + (size_t)ntiles * 512, 0, 0);  // next k-step B
    v16h a0 = load_frag_a(&tileA[buf][(wm * 32) * 32], 32, lane, 0);
    v16h a1 = load_frag_a(&tileA[buf][(wm * 32 + 16) * 32], 32, lane, 0);
#pragma unroll
    for (int nt = 0; nt < 4; ++nt) {
      v16h bf = load_frag_b_packed(btile + nt * 512, lane);
      acc[0][nt] = wmma16(a0, bf, acc[0][nt]);
      acc[1][nt] = wmma16(a1, bf, acc[1][nt]);
    }

    __syncthreads();                     // everyone done reading buf
    if (k0 + 64 < K)
      async_load_b128(buf ? lds1 : lds0, agp + k0 + 64);
  }

  // Epilogue: lane L -> n = tile_n + (L&15); VGPR r -> m = tile_m + r + 8*(L>>4).
  for (int mt = 0; mt < 2; ++mt) {
    for (int nt = 0; nt < 4; ++nt) {
      const int nn = n0 + nt * 16 + (lane & 15);
      const float bv = bias[nn];
      for (int r = 0; r < 8; ++r) {
        const int mm = m0 + mt * 16 + r + ((lane >> 4) << 3);
        float v = acc[mt][nt][r] + bv;
        if constexpr (MODE == 0) {
          int part = nn / DIMC;            // 0=q 1=k 2=v
          int cc = nn - part * DIMC;
          int head = cc >> 6;
          int c = cc & 63;
          if (part == 0) v *= ATT_SCALE;
          int b = mm / SEQ;
          int s = mm - b * SEQ;
          int y = s / WIMG, xx = s - y * WIMG;
          int win = b * NWIN + (y / WS) * NWIN_SIDE + (xx / WS);
          int pos = (y % WS) * WS + (xx % WS);
          size_t wh = (size_t)win * HEADS + head;
          half_t hv = (half_t)v;
          if (part == 0)
            ((half_t*)out0)[(wh * NKPAD + pos) * HEAD_DIM + c] = hv;
          else if (part == 1)
            ((half_t*)out1)[(wh * NKPAD + pos) * HEAD_DIM + c] = hv;
          else
            ((half_t*)out2)[(wh * HEAD_DIM + c) * NKPAD + pos] = hv;
        } else if constexpr (MODE == 2) {
          float g = 0.5f * v * (1.0f + erff(v * 0.70710678118654752f));
          ((half_t*)out0)[(size_t)mm * N + nn] = (half_t)g;
        } else {  // MODE 1 / 3: residual f32 out
          ((float*)out0)[(size_t)mm * N + nn] = v + resid[(size_t)mm * N + nn];
        }
      }
    }
  }
  (void)out1; (void)out2; (void)M;
}

// ---------------------------------------------------------------------------
// Windowed attention, one workgroup per (window, head); 8 waves own 16-query
// tiles. S epilogue stores raw scores (pad keys = -inf); rel-pos bias is
// added during the softmax streaming pass; P kept unnormalized with 1/sum
// folded into the O epilogue.
// ---------------------------------------------------------------------------
#define SMEM_SROW ((size_t)8 * 16 * NKPAD * 4)  // f32 S rows
#define SMEM_PROW ((size_t)8 * 16 * NKPAD * 2)  // f16 exp(S) rows
#define SMEM_BIAS ((size_t)8 * 16 * WS * 4)     // per bias table
#define SMEM_RED  ((size_t)8 * 32 * 4)          // partial max/sum
#define SMEM_RINV ((size_t)8 * 16 * 4)          // per-row 1/sum
#define ATTN_SMEM (SMEM_SROW + SMEM_PROW + 2 * SMEM_BIAS + SMEM_RED + SMEM_RINV)

__global__ __launch_bounds__(256) void attn_kernel(
    const half_t* __restrict__ qbuf, const half_t* __restrict__ kbuf,
    const half_t* __restrict__ vT, const float* __restrict__ rph,
    const float* __restrict__ rpw, half_t* __restrict__ out) {
  extern __shared__ char smem[];
  float*  srow_all  = (float*)smem;
  half_t* prow_all  = (half_t*)(smem + SMEM_SROW);
  float*  biash_all = (float*)(smem + SMEM_SROW + SMEM_PROW);
  float*  biasw_all = biash_all + 8 * 16 * WS;
  float*  red_all   = (float*)(smem + SMEM_SROW + SMEM_PROW + 2 * SMEM_BIAS);
  float*  rinv_all  = red_all + 8 * 32;

  const int lane = threadIdx.x & 31;
  const int wave = threadIdx.x >> 5;
  const int wh = blockIdx.x;                 // win*HEADS + head
  const int win = wh / HEADS, head = wh - win * HEADS;
  const half_t* qb = qbuf + (size_t)wh * NKPAD * HEAD_DIM;
  const half_t* kb = kbuf + (size_t)wh * NKPAD * HEAD_DIM;
  const half_t* vb = vT + (size_t)wh * HEAD_DIM * NKPAD;
  float*  srow = srow_all + (size_t)wave * 16 * NKPAD;
  half_t* prow = prow_all + (size_t)wave * 16 * NKPAD;
  float*  bh   = biash_all + wave * 16 * WS;
  float*  bw   = biasw_all + wave * 16 * WS;
  float*  redw = red_all + wave * 32;
  float*  rinw = rinv_all + wave * 16;

  const int b = win / NWIN;
  const int wloc = win - b * NWIN;
  const int wy = wloc / NWIN_SIDE, wx = wloc - wy * NWIN_SIDE;

#pragma clang loop unroll(disable)
  for (int qt = wave; qt < 13; qt += 8) {     // ceil(196/16) query tiles
    // ---- decomposed rel-pos bias: bh[m][kh] = q[m]·Rh[dist(qh,kh)] ----
#pragma clang loop unroll(disable)
    for (int j = lane; j < 16 * WS; j += 32) {
      int m = j / WS, kk = j - m * WS;
      int qpos = qt * 16 + m;
      int qh = qpos / WS; if (qh > WS - 1) qh = WS - 1;    // clamp pad rows
      int qw = qpos - (qpos / WS) * WS;
      const half_t* qrow = qb + (size_t)qpos * HEAD_DIM;
      const float* th = rph + (size_t)(qh - kk + WS - 1) * HEAD_DIM;
      const float* tw = rpw + (size_t)(qw - kk + WS - 1) * HEAD_DIM;
      float dh = 0.f, dw = 0.f;
#pragma clang loop unroll_count(8)
      for (int c = 0; c < HEAD_DIM; ++c) {
        float qv = (float)qrow[c];
        dh += qv * th[c];
        dw += qv * tw[c];
      }
      bh[m * WS + kk] = dh;
      bw[m * WS + kk] = dw;
    }

    // ---- S = q k^T into LDS f32 rows; pad keys forced to -inf ----
    const half_t* qtile = qb + (size_t)qt * 16 * HEAD_DIM;
    v16h aq0 = load_frag_a(qtile, HEAD_DIM, lane, 0);
    v16h aq1 = load_frag_a(qtile, HEAD_DIM, lane, 32);
#pragma clang loop unroll(disable)
    for (int nt = 0; nt < NKPAD / 16; ++nt) {
      v8f acc = {};
      acc = wmma16(aq0, load_frag_b_cols(kb, HEAD_DIM, lane, nt * 16, 0), acc);
      acc = wmma16(aq1, load_frag_b_cols(kb, HEAD_DIM, lane, nt * 16, 32), acc);
      int key = nt * 16 + (lane & 15);
      bool kvalid = key < NW;
#pragma unroll
      for (int r = 0; r < 8; ++r) {
        int m = r + ((lane >> 4) << 3);
        srow[m * NKPAD + key] = kvalid ? acc[r] : -3.0e38f;
      }
    }

    // ---- softmax: 2 lanes per row; bias added in the max pass (write-back),
    //      then P = exp(S - max), sum reduced via LDS ----
    {
      const int r = lane & 15, h = lane >> 4;
      float* sr = srow + r * NKPAD + h * (NKPAD / 2);
      const float* bhr = bh + r * WS;
      const float* bwr = bw + r * WS;
      float mx = -3.0e38f;
#pragma clang loop unroll(disable)
      for (int kk = 0; kk < NKPAD / 2; ++kk) {
        int key = h * (NKPAD / 2) + kk;
        int kh = key / WS;
        int kw = key - kh * WS;
        int khc = kh > WS - 1 ? WS - 1 : kh;
        float bsum = bhr[khc] + bwr[kw];          // always loaded (no branch)
        float v = sr[kk] + ((key < NW) ? bsum : 0.f);
        sr[kk] = v;                                // -inf rows stay ~-inf
        mx = v > mx ? v : mx;
      }
      redw[lane] = mx;
      float m2 = fmaxf(redw[r], redw[16 + r]);
      half_t* pr = prow + r * NKPAD + h * (NKPAD / 2);
      float sum = 0.f;
#pragma clang loop unroll_count(4)
      for (int kk = 0; kk < NKPAD / 2; ++kk) {
        float e = __expf(sr[kk] - m2);
        pr[kk] = (half_t)e;
        sum += e;
      }
      redw[lane] = sum;
      if (lane < 16) rinw[lane] = 1.0f / (redw[lane] + redw[16 + lane]);
    }

    // ---- O = P * V (vT is [c][key], key-contiguous per lane) ----
    for (int ct = 0; ct < HEAD_DIM / 16; ++ct) {
      v8f o = {};
#pragma clang loop unroll(disable)
      for (int k0 = 0; k0 < NKPAD; k0 += 32) {
        v16h a = load_frag_a(prow, NKPAD, lane, k0);
        v16h bf = load_frag_b_cols(vb, NKPAD, lane, ct * 16, k0);
        o = wmma16(a, bf, o);
      }
      int c = ct * 16 + (lane & 15);
#pragma unroll
      for (int r = 0; r < 8; ++r) {
        int m = r + ((lane >> 4) << 3);
        int qpos = qt * 16 + m;
        if (qpos < NW) {
          int py = qpos / WS, px = qpos - py * WS;
          int y = wy * WS + py, xx = wx * WS + px;
          size_t row = (size_t)b * SEQ + (size_t)y * WIMG + xx;
          out[row * DIMC + head * HEAD_DIM + c] = (half_t)(o[r] * rinw[m]);
        }
      }
    }
  }
}

// ---------------------------------------------------------------------------
// Host orchestration
// ---------------------------------------------------------------------------
extern "C" void kernel_launch(void* const* d_in, const int* in_sizes, int n_in,
                              void* d_out, int out_size, void* d_ws, size_t ws_size,
                              hipStream_t stream) {
  const float* x     = (const float*)d_in[0];
  const float* ln1s  = (const float*)d_in[1];
  const float* ln1b  = (const float*)d_in[2];
  const float* qkvw  = (const float*)d_in[3];
  const float* qkvb  = (const float*)d_in[4];
  const float* rph   = (const float*)d_in[5];
  const float* rpw   = (const float*)d_in[6];
  const float* projw = (const float*)d_in[7];
  const float* projb = (const float*)d_in[8];
  const float* ln2s  = (const float*)d_in[9];
  const float* ln2b  = (const float*)d_in[10];
  const float* fc1w  = (const float*)d_in[11];
  const float* fc1b  = (const float*)d_in[12];
  const float* fc2w  = (const float*)d_in[13];
  const float* fc2b  = (const float*)d_in[14];
  float* outp = (float*)d_out;
  (void)in_sizes; (void)n_in; (void)out_size; (void)ws_size;

  char* ws = (char*)d_ws;
  size_t off = 0;
  auto alloc = [&](size_t bytes) -> char* {
    char* p = ws + off;
    off += (bytes + 255) & ~(size_t)255;
    return p;
  };
  half_t* xn   = (half_t*)alloc((size_t)MROWS * DIMC * 2);       // LN1 out f16
  half_t* wq   = (half_t*)alloc((size_t)DIMC * 3 * DIMC * 2);    // packed qkv_w
  half_t* wp   = (half_t*)alloc((size_t)DIMC * DIMC * 2);        // packed proj_w
  half_t* w1   = (half_t*)alloc((size_t)DIMC * MLP_HID * 2);     // packed fc1_w
  half_t* w2   = (half_t*)alloc((size_t)MLP_HID * DIMC * 2);     // packed fc2_w
  half_t* qbuf = (half_t*)alloc((size_t)BW * HEADS * NKPAD * HEAD_DIM * 2);
  half_t* kbuf = (half_t*)alloc((size_t)BW * HEADS * NKPAD * HEAD_DIM * 2);
  half_t* vtb  = (half_t*)alloc((size_t)BW * HEADS * HEAD_DIM * NKPAD * 2);
  half_t* aout = (half_t*)alloc((size_t)MROWS * DIMC * 2);       // attention out f16
  float*  x1   = (float*)alloc((size_t)MROWS * DIMC * 4);        // x + attn (f32)
  half_t* h2   = (half_t*)alloc((size_t)MROWS * DIMC * 2);       // LN2 out f16
  half_t* hbuf = (half_t*)alloc((size_t)MROWS * MLP_HID * 2);    // gelu(fc1) f16

  // 1) Pack all weight matrices into WMMA B-fragment layout (f16).
  pack_b<<<(DIMC * 3 * DIMC + 255) / 256, 256, 0, stream>>>(qkvw, wq, DIMC, 3 * DIMC);
  pack_b<<<(DIMC * DIMC + 255) / 256, 256, 0, stream>>>(projw, wp, DIMC, DIMC);
  pack_b<<<(DIMC * MLP_HID + 255) / 256, 256, 0, stream>>>(fc1w, w1, DIMC, MLP_HID);
  pack_b<<<(MLP_HID * DIMC + 255) / 256, 256, 0, stream>>>(fc2w, w2, MLP_HID, DIMC);

  // 2) LN1
  ln_kernel<<<MROWS, 256, 0, stream>>>(x, ln1s, ln1b, xn);

  // 3) QKV GEMM + scatter into window-major q/k/vT f16 buffers
  gemm_ep<0><<<dim3(3 * DIMC / 256, MROWS / 64), 256, 0, stream>>>(
      xn, wq, qkvb, nullptr, qbuf, kbuf, vtb, MROWS, 3 * DIMC, DIMC);

  // 4) Windowed attention (WMMA QK^T -> softmax -> WMMA PV)
  attn_kernel<<<BW * HEADS, 256, ATTN_SMEM, stream>>>(qbuf, kbuf, vtb, rph, rpw, aout);

  // 5) Proj GEMM + residual -> x1 (f32)
  gemm_ep<1><<<dim3(DIMC / 256, MROWS / 64), 256, 0, stream>>>(
      aout, wp, projb, x, x1, nullptr, nullptr, MROWS, DIMC, DIMC);

  // 6) LN2
  ln_kernel<<<MROWS, 256, 0, stream>>>(x1, ln2s, ln2b, h2);

  // 7) FC1 GEMM + exact GELU -> f16
  gemm_ep<2><<<dim3(MLP_HID / 256, MROWS / 64), 256, 0, stream>>>(
      h2, w1, fc1b, nullptr, hbuf, nullptr, nullptr, MROWS, MLP_HID, DIMC);

  // 8) FC2 GEMM + residual -> d_out (f32)
  gemm_ep<3><<<dim3(DIMC / 256, MROWS / 64), 256, 0, stream>>>(
      hbuf, w2, fc2b, x1, outp, nullptr, nullptr, MROWS, DIMC, MLP_HID);
}
